// GPT_1254130450953
// MI455X (gfx1250) — matrix operand order repaired and checked by reading, not compiled, not run on hardware
//
#include <hip/hip_runtime.h>
#include <hip/hip_bf16.h>

typedef __attribute__((ext_vector_type(16))) _Float16 v16h;
typedef __attribute__((ext_vector_type(8)))  _Float16 v8h;
typedef __attribute__((ext_vector_type(8)))  float    v8f;

constexpr int B_ = 128, T_ = 256, C_ = 384, H_ = 4, HS_ = 64, L_ = 6, V_ = 65;
constexpr int FF_ = 4 * C_;                 // 1536
constexpr int BT_ = B_ * T_;                // 32768
constexpr int QKV_ = 3 * H_ * HS_;          // 768

__device__ inline v8f wmma_f16(v16h a, v16h b, v8f c) {
  return __builtin_amdgcn_wmma_f32_16x16x32_f16(false, a, false, b, (short)0, c, false, false);
}
__device__ inline v8f vzero8() {
  v8f z;
#pragma unroll
  for (int i = 0; i < 8; ++i) z[i] = 0.f;
  return z;
}
__device__ inline v16h frag_cat(v8h lo, v8h hi) {
  return __builtin_shufflevector(lo, hi, 0, 1, 2, 3, 4, 5, 6, 7, 8, 9, 10, 11, 12, 13, 14, 15);
}
__device__ inline v16h frag_ld(const _Float16* p0, const _Float16* p1) {
  return frag_cat(*(const v8h*)p0, *(const v8h*)p1);
}
// CDNA5 async global->LDS copy of 16 bytes per lane (ASYNCcnt-tracked).
// Generic pointers to LDS carry the wave-relative LDS offset in the low 32 bits.
__device__ inline void async_ld16(void* lds, const void* g) {
  asm volatile("global_load_async_to_lds_b128 %0, %1, off"
               :: "v"((unsigned)(unsigned long long)lds), "v"(g) : "memory");
}
__device__ inline void wait_async0() {
  asm volatile("s_wait_asynccnt 0" ::: "memory");
}

// ---------------------------------------------------------------- utilities
__global__ void k_zero(float* p) {
  if (threadIdx.x == 0 && blockIdx.x == 0) *p = 0.f;
}

// transpose+convert: src [Ld][K][N] f32 -> dst [Ld][N][K] f16
// grid: (ceil(K/256), N, Ld) -- division-free
__global__ void k_cvt_t(const float* __restrict__ src, _Float16* __restrict__ dst,
                        int K, int N) {
  int k = blockIdx.x * 256 + threadIdx.x;
  if (k >= K) return;
  int n = blockIdx.y, l = blockIdx.z;
  dst[((long)l * N + n) * K + k] = (_Float16)src[((long)l * K + k) * N + n];
}

// Pack Wq/Wk/Wv (L,H,C,HS) -> per-layer TRANSPOSED f16 [768][C]
// grid: (ceil(C/256), 768, L) -- division-free
__global__ void k_pack_qkv(const float* __restrict__ Wq, const float* __restrict__ Wk,
                           const float* __restrict__ Wv, _Float16* __restrict__ out) {
  int c = blockIdx.x * 256 + threadIdx.x;
  if (c >= C_) return;
  int n = blockIdx.y, l = blockIdx.z;
  int sel = n >> 8, h = (n >> 6) & 3, d = n & 63;
  const float* W = (sel == 0) ? Wq : (sel == 1) ? Wk : Wv;
  out[((long)l * QKV_ + n) * C_ + c] = (_Float16)W[(((long)l * H_ + h) * C_ + c) * HS_ + d];
}

__global__ void k_embed(const int* __restrict__ idx, const float* __restrict__ tok,
                        const float* __restrict__ pos, float* __restrict__ x, int n) {
  int i = blockIdx.x * blockDim.x + threadIdx.x;
  if (i >= n) return;
  int t = i / C_;
  int c = i % C_;
  x[i] = tok[(long)idx[t] * C_ + c] + pos[(long)(t % T_) * C_ + c];
}

// One wave per token LayerNorm over C=384; output f16 (feeds WMMA GEMMs only)
__global__ void __launch_bounds__(256) k_ln(const float* __restrict__ x,
                                            const float* __restrict__ g,
                                            const float* __restrict__ b,
                                            _Float16* __restrict__ y) {
  int w = threadIdx.x >> 5, lane = threadIdx.x & 31;
  long tok = (long)blockIdx.x * 8 + w;
  const float* xr = x + tok * C_;
  float v[12];
  float s = 0.f;
#pragma unroll
  for (int i = 0; i < 12; ++i) { v[i] = xr[lane + i * 32]; s += v[i]; }
#pragma unroll
  for (int m = 16; m >= 1; m >>= 1) s += __shfl_xor(s, m, 32);
  float mu = s * (1.0f / C_);
  float var = 0.f;
#pragma unroll
  for (int i = 0; i < 12; ++i) { float d = v[i] - mu; var += d * d; }
#pragma unroll
  for (int m = 16; m >= 1; m >>= 1) var += __shfl_xor(var, m, 32);
  float r = rsqrtf(var * (1.0f / C_) + 1e-5f);
#pragma unroll
  for (int i = 0; i < 12; ++i) {
    int c = lane + i * 32;
    y[tok * C_ + c] = (_Float16)((v[i] - mu) * r * g[c] + b[c]);
  }
}

// ---------------------------------------------------------------- WMMA GEMM
// epi 0: OutH = A*Wt^T + 0          (f16 out)
// epi 1: OutH = relu(A*Wt^T + bias) (f16 out)
// epi 2: OutF = res + A*Wt^T + bias (f32 out, residual)
// A: [M][K] f16.  Wt: TRANSPOSED weights [N][K] f16.
// Block 256 threads = 8 waves; tile 128(M) x 64(N); K-step 32.
// Double-buffered LDS: async DMA of tile i+1 overlaps WMMA on tile i.
__global__ void __launch_bounds__(256) k_gemm(const _Float16* __restrict__ A,
                                              const _Float16* __restrict__ Wt,
                                              const float* __restrict__ bias,
                                              const float* __restrict__ res,
                                              float* __restrict__ OutF,
                                              _Float16* __restrict__ OutH,
                                              int N, int K, int epi) {
  __shared__ alignas(16) _Float16 As[2][128 * 32];
  __shared__ alignas(16) _Float16 Bs[2][64 * 32];
  int tid = threadIdx.x, lane = tid & 31, w = tid >> 5;
  int hl = lane >> 4, l16 = lane & 15;
  long rowBase = (long)blockIdx.y * 128;
  int colBase = blockIdx.x * 64;

  auto stage = [&](int buf, int k0) {
#pragma unroll
    for (int i = 0; i < 2; ++i) {                 // A tile 128x32 f16 (8KB)
      int chunk = tid + i * 256;
      int r = chunk >> 2, c = (chunk & 3) * 8;
      async_ld16(&As[buf][r * 32 + c], A + (rowBase + r) * K + k0 + c);
    }
    int r = tid >> 2, c = (tid & 3) * 8;          // B tile 64x32 f16 (4KB)
    async_ld16(&Bs[buf][r * 32 + c], Wt + (long)(colBase + r) * K + k0 + c);
  };

  v8f acc[4];
#pragma unroll
  for (int i = 0; i < 4; ++i) acc[i] = vzero8();

  int nk = K >> 5;
  stage(0, 0);
  int buf = 0;
  for (int i = 0; i < nk; ++i, buf ^= 1) {
    wait_async0();                                // tile i landed in LDS
    __syncthreads();                              // visible to all; prev reads done
    if (i + 1 < nk) stage(buf ^ 1, (i + 1) * 32); // DMA next tile during compute

    // batch all fragment loads before the WMMA chain (staggered dscnt waits)
    const _Float16* ap = &As[buf][(w * 16 + l16) * 32];
    v16h a = frag_ld(ap + hl * 8, ap + 16 + hl * 8);
    v16h bfr[4];
#pragma unroll
    for (int nt = 0; nt < 4; ++nt) {
      const _Float16* bp = &Bs[buf][(nt * 16 + l16) * 32 + hl * 8];
      bfr[nt] = frag_ld(bp, bp + 16);
    }
#pragma unroll
    for (int nt = 0; nt < 4; ++nt) acc[nt] = wmma_f16(a, bfr[nt], acc[nt]);
  }
  // Epilogue: D layout M = r + 8*hl, N = l16
#pragma unroll
  for (int nt = 0; nt < 4; ++nt) {
#pragma unroll
    for (int r = 0; r < 8; ++r) {
      long row = rowBase + w * 16 + r + 8 * hl;
      int col = colBase + nt * 16 + l16;
      float val = acc[nt][r] + (bias ? bias[col] : 0.f);
      if (epi == 2)      OutF[row * N + col] = val + res[row * N + col];
      else if (epi == 1) OutH[row * N + col] = (_Float16)fmaxf(val, 0.f);
      else               OutH[row * N + col] = (_Float16)val;
    }
  }
}

// ---------------------------------------------------------------- attention
// One block per (b,h). Flash-style online softmax; K/V streamed in 32-col chunks.
// qkv is f16 [token][768] = [q|k|v], within-part col = h*64+d.
// kh [seq32][d64] via async copy; vh TRANSPOSED [d64][seq32]. Double-buffered.
__global__ void __launch_bounds__(256) k_attn(const _Float16* __restrict__ qkv,
                                              _Float16* __restrict__ att) {
  __shared__ alignas(16) _Float16 kh[2][32 * 64];
  __shared__ alignas(16) _Float16 vh[2][64 * 32];
  __shared__ alignas(16) _Float16 ps[8 * 16 * 32];   // per-wave P restage
  int bh = blockIdx.x;
  int b = bh / H_, h = bh % H_;
  int tid = threadIdx.x, lane = tid & 31, w = tid >> 5;
  int hl = lane >> 4, l16 = lane & 15;
  const _Float16 qs = (_Float16)0.125f;           // 1/sqrt(64), exact in f16

  auto stage = [&](int bufi, int j0) {
    int r = tid >> 2, d = (tid & 3) * 8;
    long base = ((long)(b * T_ + j0 + r)) * QKV_ + h * HS_ + d;
    async_ld16(&kh[bufi][r * 64 + d], qkv + base + 256);
    v8h vv = *(const v8h*)(qkv + base + 512);
#pragma unroll
    for (int j = 0; j < 8; ++j) vh[bufi][(d + j) * 32 + r] = vv[j];
  };

  v16h qf[2][2];
  v8f o[2][4];
  float mrow[2][8], lrow[2][8];
#pragma unroll
  for (int t2 = 0; t2 < 2; ++t2) {
    int it = w + t2 * 8;
    const _Float16* qp0 = qkv + ((long)(b * T_ + it * 16 + l16)) * QKV_ + h * HS_;
#pragma unroll
    for (int kh_i = 0; kh_i < 2; ++kh_i) {
      const _Float16* qp = qp0 + kh_i * 32;
      v8h lo = *(const v8h*)(qp + hl * 8);
      v8h hi = *(const v8h*)(qp + 16 + hl * 8);
#pragma unroll
      for (int j = 0; j < 8; ++j) { lo[j] = lo[j] * qs; hi[j] = hi[j] * qs; }
      qf[t2][kh_i] = frag_cat(lo, hi);
    }
#pragma unroll
    for (int nt = 0; nt < 4; ++nt) o[t2][nt] = vzero8();
#pragma unroll
    for (int r = 0; r < 8; ++r) { mrow[t2][r] = -1e30f; lrow[t2][r] = 0.f; }
  }

  stage(0, 0);
  int buf = 0;
  for (int j0 = 0; j0 < T_; j0 += 32, buf ^= 1) {
    wait_async0();
    __syncthreads();
    if (j0 + 32 < T_) stage(buf ^ 1, j0 + 32);    // overlap next chunk with compute

#pragma unroll
    for (int t2 = 0; t2 < 2; ++t2) {
      int it = w + t2 * 8;
      if (j0 <= it * 16 + 15) {                   // wave-uniform causal skip
        // batch all 4 K-fragments, then 4 S-WMMAs
        v16h kf[2][2];
#pragma unroll
        for (int sub = 0; sub < 2; ++sub)
#pragma unroll
          for (int dh = 0; dh < 2; ++dh) {
            const _Float16* kp = &kh[buf][(sub * 16 + l16) * 64 + dh * 32 + hl * 8];
            kf[sub][dh] = frag_ld(kp, kp + 16);
          }
        v8f sa[2];
#pragma unroll
        for (int sub = 0; sub < 2; ++sub) {       // S = Q(16x64) . K^T(64x16)
          sa[sub] = vzero8();
#pragma unroll
          for (int dh = 0; dh < 2; ++dh)
            sa[sub] = wmma_f16(qf[t2][dh], kf[sub][dh], sa[sub]);
        }
        _Float16* pw = ps + w * 512;
#pragma unroll
        for (int r = 0; r < 8; ++r) {             // mask + online softmax per row
          int rowg = it * 16 + r + 8 * hl;
          float s0 = (j0 + l16 <= rowg)      ? sa[0][r] : -1e30f;
          float s1 = (j0 + 16 + l16 <= rowg) ? sa[1][r] : -1e30f;
          float mx = fmaxf(s0, s1);
#pragma unroll
          for (int m = 8; m >= 1; m >>= 1) mx = fmaxf(mx, __shfl_xor(mx, m, 32));
          float mnew = fmaxf(mrow[t2][r], mx);
          float corr = __expf(mrow[t2][r] - mnew);
          float p0 = __expf(s0 - mnew), p1 = __expf(s1 - mnew);
          float psum = p0 + p1;
#pragma unroll
          for (int m = 8; m >= 1; m >>= 1) psum += __shfl_xor(psum, m, 32);
          lrow[t2][r] = lrow[t2][r] * corr + psum;
          mrow[t2][r] = mnew;
#pragma unroll
          for (int nt = 0; nt < 4; ++nt) o[t2][nt][r] *= corr;
          pw[(r + 8 * hl) * 32 + l16]      = (_Float16)p0;
          pw[(r + 8 * hl) * 32 + 16 + l16] = (_Float16)p1;
        }
        // reload P in A-frag layout + batch all 4 V-fragments, then 4 O-WMMAs
        const _Float16* pp = pw + l16 * 32;
        v16h pa = frag_ld(pp + hl * 8, pp + 16 + hl * 8);
        v16h vf[4];
#pragma unroll
        for (int nt = 0; nt < 4; ++nt) {
          const _Float16* vp = &vh[buf][(nt * 16 + l16) * 32 + hl * 8];
          vf[nt] = frag_ld(vp, vp + 16);
        }
#pragma unroll
        for (int nt = 0; nt < 4; ++nt)            // O += P(16x32) . V(32x64)
          o[t2][nt] = wmma_f16(pa, vf[nt], o[t2][nt]);
      }
    }
  }
#pragma unroll
  for (int t2 = 0; t2 < 2; ++t2) {
    int it = w + t2 * 8;
#pragma unroll
    for (int r = 0; r < 8; ++r) {
      int rowg = it * 16 + r + 8 * hl;
      float inv = 1.0f / lrow[t2][r];
      long obase = ((long)(b * T_ + rowg)) * (H_ * HS_) + h * HS_;
#pragma unroll
      for (int nt = 0; nt < 4; ++nt)
        att[obase + nt * 16 + l16] = (_Float16)(o[t2][nt][r] * inv);
    }
  }
}

// ---------------------------------------------- final: lnf + LM head + loss
__global__ void __launch_bounds__(256) k_final(const float* __restrict__ x,
                                               const float* __restrict__ g,
                                               const float* __restrict__ bb,
                                               const float* __restrict__ Wlm,
                                               const float* __restrict__ blm,
                                               const int* __restrict__ targets,
                                               float* __restrict__ logits,
                                               float* __restrict__ loss) {
  __shared__ float xs[8 * C_];
  __shared__ float ls[8 * 80];
  int w = threadIdx.x >> 5, lane = threadIdx.x & 31;
  long tok = (long)blockIdx.x * 8 + w;
  const float* xr = x + tok * C_;
  float v[12];
  float s = 0.f;
#pragma unroll
  for (int i = 0; i < 12; ++i) { v[i] = xr[lane + i * 32]; s += v[i]; }
#pragma unroll
  for (int m = 16; m >= 1; m >>= 1) s += __shfl_xor(s, m, 32);
  float mu = s * (1.0f / C_);
  float var = 0.f;
#pragma unroll
  for (int i = 0; i < 12; ++i) { float d = v[i] - mu; var += d * d; }
#pragma unroll
  for (int m = 16; m >= 1; m >>= 1) var += __shfl_xor(var, m, 32);
  float r = rsqrtf(var * (1.0f / C_) + 1e-5f);
#pragma unroll
  for (int i = 0; i < 12; ++i) {
    int c = lane + i * 32;
    xs[w * C_ + c] = (v[i] - mu) * r * g[c] + bb[c];
  }
  float lg[3];
#pragma unroll
  for (int i = 0; i < 3; ++i) {
    int vv = lane + i * 32;
    float a = -1e30f;
    if (vv < V_) {
      a = blm[vv];
      for (int c = 0; c < C_; ++c) a += xs[w * C_ + c] * Wlm[(long)c * V_ + vv];
      logits[tok * V_ + vv] = a;
      ls[w * 80 + vv] = a;
    }
    lg[i] = a;
  }
  float mx = fmaxf(fmaxf(lg[0], lg[1]), lg[2]);
#pragma unroll
  for (int m = 16; m >= 1; m >>= 1) mx = fmaxf(mx, __shfl_xor(mx, m, 32));
  float se = __expf(lg[0] - mx) + __expf(lg[1] - mx) + __expf(lg[2] - mx);
#pragma unroll
  for (int m = 16; m >= 1; m >>= 1) se += __shfl_xor(se, m, 32);
  float lse = mx + __logf(se);
  if (lane == 0) {
    int t = targets[tok];
    float lp = ls[w * 80 + t] - lse;
    atomicAdd(loss, -lp / (float)BT_);
  }
}

// ---------------------------------------------------------------- host side
extern "C" void kernel_launch(void* const* d_in, const int* in_sizes, int n_in,
                              void* d_out, int out_size, void* d_ws, size_t ws_size,
                              hipStream_t stream) {
  (void)in_sizes; (void)n_in; (void)out_size; (void)ws_size;
  const int*   idx     = (const int*)d_in[0];
  const int*   targets = (const int*)d_in[1];
  const float* tok_emb = (const float*)d_in[2];
  const float* pos_emb = (const float*)d_in[3];
  const float* Wq      = (const float*)d_in[4];
  const float* Wk      = (const float*)d_in[5];
  const float* Wv      = (const float*)d_in[6];
  const float* Wo      = (const float*)d_in[7];
  const float* bo      = (const float*)d_in[8];
  const float* W1      = (const float*)d_in[9];
  const float* b1      = (const float*)d_in[10];
  const float* W2      = (const float*)d_in[11];
  const float* b2      = (const float*)d_in[12];
  const float* ln1_g   = (const float*)d_in[13];
  const float* ln1_b   = (const float*)d_in[14];
  const float* ln2_g   = (const float*)d_in[15];
  const float* ln2_b   = (const float*)d_in[16];
  const float* lnf_g   = (const float*)d_in[17];
  const float* lnf_b   = (const float*)d_in[18];
  const float* Wlm     = (const float*)d_in[19];
  const float* blm     = (const float*)d_in[20];

  float* logits = (float*)d_out;
  float* loss   = logits + (long)BT_ * V_;

  // workspace carve-out (256B aligned)
  char* ws = (char*)d_ws;
  size_t off = 0;
  auto carve = [&](size_t bytes) { char* p = ws + off; off = (off + bytes + 255) & ~(size_t)255; return p; };
  float*    x      = (float*)   carve((size_t)BT_ * C_ * 4);       // residual stream (f32)
  _Float16* xnh    = (_Float16*)carve((size_t)BT_ * C_ * 2);       // LN output
  _Float16* qkvh   = (_Float16*)carve((size_t)BT_ * QKV_ * 2);
  _Float16* atth   = (_Float16*)carve((size_t)BT_ * H_ * HS_ * 2);
  _Float16* ffh    = (_Float16*)carve((size_t)BT_ * FF_ * 2);
  _Float16* wqkv_t = (_Float16*)carve((size_t)L_ * QKV_ * C_ * 2); // [768][C]
  _Float16* wo_t   = (_Float16*)carve((size_t)L_ * C_ * (H_ * HS_) * 2); // [C][256]
  _Float16* w1_t   = (_Float16*)carve((size_t)L_ * FF_ * C_ * 2);  // [FF][C]
  _Float16* w2_t   = (_Float16*)carve((size_t)L_ * C_ * FF_ * 2);  // [C][FF]

  k_zero<<<1, 1, 0, stream>>>(loss);

  k_pack_qkv<<<dim3((C_ + 255) / 256, QKV_, L_), 256, 0, stream>>>(Wq, Wk, Wv, wqkv_t);
  k_cvt_t<<<dim3((H_ * HS_ + 255) / 256, C_, L_), 256, 0, stream>>>(Wo, wo_t, H_ * HS_, C_);
  k_cvt_t<<<dim3((C_ + 255) / 256, FF_, L_), 256, 0, stream>>>(W1, w1_t, C_, FF_);
  k_cvt_t<<<dim3((FF_ + 255) / 256, C_, L_), 256, 0, stream>>>(W2, w2_t, FF_, C_);

  int nx = BT_ * C_;
  k_embed<<<(nx + 255) / 256, 256, 0, stream>>>(idx, tok_emb, pos_emb, x, nx);

  dim3 gQKV(QKV_ / 64, BT_ / 128);
  dim3 gC(C_ / 64, BT_ / 128);
  dim3 gFF(FF_ / 64, BT_ / 128);

  for (int l = 0; l < L_; ++l) {
    k_ln<<<BT_ / 8, 256, 0, stream>>>(x, ln1_g + l * C_, ln1_b + l * C_, xnh);
    k_gemm<<<gQKV, 256, 0, stream>>>(xnh, wqkv_t + (size_t)l * QKV_ * C_,
                                     nullptr, nullptr, nullptr, qkvh, QKV_, C_, 0);
    k_attn<<<B_ * H_, 256, 0, stream>>>(qkvh, atth);
    k_gemm<<<gC, 256, 0, stream>>>(atth, wo_t + (size_t)l * C_ * (H_ * HS_),
                                   bo + l * C_, x, x, nullptr, C_, H_ * HS_, 2);
    k_ln<<<BT_ / 8, 256, 0, stream>>>(x, ln2_g + l * C_, ln2_b + l * C_, xnh);
    k_gemm<<<gFF, 256, 0, stream>>>(xnh, w1_t + (size_t)l * FF_ * C_,
                                    b1 + l * FF_, nullptr, nullptr, ffh, FF_, C_, 1);
    k_gemm<<<gC, 256, 0, stream>>>(ffh, w2_t + (size_t)l * C_ * FF_,
                                   b2 + l * C_, x, x, nullptr, C_, FF_, 2);
  }
  k_final<<<BT_ / 8, 256, 0, stream>>>(x, lnf_g, lnf_b, Wlm, blm, targets, logits, loss);
}
